// LSTMModel_5471788335350
// MI455X (gfx1250) — matrix operand order repaired and checked by reading, not compiled, not run on hardware
//
#include <hip/hip_runtime.h>
#include <hip/hip_bf16.h>

#define UNITS 1024
#define EMBD  512
#define SEQL  256
#define BATCH 256

typedef unsigned short u16;
typedef __attribute__((ext_vector_type(16))) __bf16 v16bf;
typedef __attribute__((ext_vector_type(8)))  float  v8f;

union BfFrag { v16bf v; uint4 q[2]; };

__device__ __forceinline__ unsigned pack2_bf16(float lo, float hi) {
  unsigned a = __float_as_uint(lo);
  unsigned b = __float_as_uint(hi);
  a += 0x7FFFu + ((a >> 16) & 1u);   // round-to-nearest-even
  b += 0x7FFFu + ((b >> 16) & 1u);
  return (a >> 16) | (b & 0xFFFF0000u);
}
__device__ __forceinline__ u16 f2bf(float x) {
  unsigned u = __float_as_uint(x);
  u += 0x7FFFu + ((u >> 16) & 1u);
  return (u16)(u >> 16);
}
__device__ __forceinline__ float bf2f(u16 v) {
  return __uint_as_float(((unsigned)v) << 16);
}
__device__ __forceinline__ float sigm(float x) { return 1.0f / (1.0f + __expf(-x)); }

// ---------------------------------------------------------------------------
// One-time: W [K,N] f32 row-major -> WT [N,K] bf16 (B-fragment K contiguous)
// ---------------------------------------------------------------------------
__global__ void transpose_bf16_kernel(const float* __restrict__ W,
                                      u16* __restrict__ WT, int K, int N) {
  const int idx = blockIdx.x * blockDim.x + threadIdx.x;
  if (idx >= K * N) return;
  const int k = idx / N;
  const int n = idx - k * N;
  WT[(size_t)n * K + k] = f2bf(W[idx]);
}

// One-time: embedding table f32 -> bf16 (same layout), 51.2 MB, L2-resident.
__global__ void convert_bf16_kernel(const float* __restrict__ src,
                                    u16* __restrict__ dst, int n) {
  const int i = blockIdx.x * blockDim.x + threadIdx.x;
  if (i < n) dst[i] = f2bf(src[i]);
}

__global__ void init_state_kernel(float* __restrict__ c, u16* __restrict__ hb) {
  const int i = blockIdx.x * blockDim.x + threadIdx.x;  // exactly BATCH*UNITS
  c[i]  = 0.0f;
  hb[i] = 0;
}

// ---------------------------------------------------------------------------
// Fused LSTM step:  gates = emb[x_t]@Wx + h@Wh + b ; cell update ; h out.
// grid = 128 (8 M-blocks x 16 U-blocks), block = 256 (8 waves).
// Wave tile: 16(M) x 16(U), 4 gate accumulators (v_wmma_f32_16x16x32_bf16).
// h double-buffered across steps. PRECONV: embedding table already in bf16,
// so every A/B fragment is two straight global_load_b128 (no VALU packing).
// ---------------------------------------------------------------------------
template <bool PRECONV>
__global__ __launch_bounds__(256) void lstm_step_kernel(
    const int*   __restrict__ sents,   // [BATCH, SEQL]
    const float* __restrict__ embf,    // [VOCAB, EMBD] f32 (fallback path)
    const u16*   __restrict__ embb,    // [VOCAB, EMBD] bf16 (preconv path)
    const u16*   __restrict__ WxT,     // [4*UNITS, EMBD] bf16
    const u16*   __restrict__ WhT,     // [4*UNITS, UNITS] bf16
    const float* __restrict__ bias,    // [4*UNITS]
    float*       __restrict__ cst,     // [BATCH, UNITS] f32 (owned per tile)
    const u16*   __restrict__ hbin,    // [BATCH, UNITS] bf16 (h_{t-1})
    u16*         __restrict__ hbout,   // [BATCH, UNITS] bf16 (h_t)
    int t)
{
  const int lane  = threadIdx.x & 31;
  const int wave  = threadIdx.x >> 5;
  const int mt    = wave >> 2;           // 0..1
  const int nt    = wave & 3;            // 0..3
  const int bx    = blockIdx.x & 7;      // 8 M-blocks of 32 rows
  const int by    = blockIdx.x >> 3;     // 16 U-blocks of 64 units
  const int m0    = bx * 32 + mt * 16;
  const int u0    = by * 64 + nt * 16;
  const int lhalf = lane >> 4;           // 0: lanes 0-15, 1: lanes 16-31
  const int lmod  = lane & 15;
  const int arow  = m0 + lmod;           // A-matrix row handled by this lane
  const int bcol  = u0 + lmod;           // gate-local unit column for B

  const v8f vzero = {0.f, 0.f, 0.f, 0.f, 0.f, 0.f, 0.f, 0.f};
  v8f acc[4] = {vzero, vzero, vzero, vzero};   // i, f, g, o

  const int token = sents[arow * SEQL + t];

  // ---- x-part: K = 0..511 over emb[x_t] @ Wx --------------------------------
  for (int kc = 0; kc < EMBD / 32; ++kc) {
    // A fragment (16-bit A 16x32 layout): half 0 -> K {0..7,16..23}, half 1 -> +8
    const int kbA = kc * 32 + lhalf * 8;
    BfFrag a;
    if (PRECONV) {
      const uint4* ap = (const uint4*)(embb + (size_t)token * EMBD + kbA);
      a.q[0] = ap[0];   // K kbA .. kbA+7
      a.q[1] = ap[2];   // K kbA+16 .. kbA+23
    } else {
      const float* xrow = embf + (size_t)token * EMBD;
      const float4 x0 = *(const float4*)(xrow + kbA);
      const float4 x1 = *(const float4*)(xrow + kbA + 4);
      const float4 x2 = *(const float4*)(xrow + kbA + 16);
      const float4 x3 = *(const float4*)(xrow + kbA + 20);
      a.q[0] = make_uint4(pack2_bf16(x0.x, x0.y), pack2_bf16(x0.z, x0.w),
                          pack2_bf16(x1.x, x1.y), pack2_bf16(x1.z, x1.w));
      a.q[1] = make_uint4(pack2_bf16(x2.x, x2.y), pack2_bf16(x2.z, x2.w),
                          pack2_bf16(x3.x, x3.y), pack2_bf16(x3.z, x3.w));
    }
    // B fragment (32x16): half 0 -> K {0..7,16..23}, half 1 -> +8 (matches A)
    const int kbB = kc * 32 + lhalf * 8;
#pragma unroll
    for (int g = 0; g < 4; ++g) {
      BfFrag bfr;
      const uint4* bp = (const uint4*)(WxT + (size_t)(g * UNITS + bcol) * EMBD + kbB);
      bfr.q[0] = bp[0];
      bfr.q[1] = bp[2];
      acc[g] = __builtin_amdgcn_wmma_f32_16x16x32_bf16(
          false, a.v, false, bfr.v, (short)0, acc[g], false, false);
    }
  }

  // ---- h-part: K = 0..1023 over h_{t-1} @ Wh --------------------------------
  for (int kc = 0; kc < UNITS / 32; ++kc) {
    const int kbA = kc * 32 + lhalf * 8;
    BfFrag a;
    const uint4* ap = (const uint4*)(hbin + (size_t)arow * UNITS + kbA);
    a.q[0] = ap[0];
    a.q[1] = ap[2];
    const int kbB = kc * 32 + lhalf * 8;
#pragma unroll
    for (int g = 0; g < 4; ++g) {
      BfFrag bfr;
      const uint4* bp = (const uint4*)(WhT + (size_t)(g * UNITS + bcol) * UNITS + kbB);
      bfr.q[0] = bp[0];
      bfr.q[1] = bp[2];
      acc[g] = __builtin_amdgcn_wmma_f32_16x16x32_bf16(
          false, a.v, false, bfr.v, (short)0, acc[g], false, false);
    }
  }

  // ---- fused bias + activations + cell update -------------------------------
  // C/D layout: vgpr r, lane l -> row m0 + r + 8*(l>=16), col u0 + (l&15)
  const int   ucol = u0 + lmod;
  const float bi   = bias[ucol];
  const float bff  = bias[UNITS + ucol];
  const float bg   = bias[2 * UNITS + ucol];
  const float bo   = bias[3 * UNITS + ucol];

#pragma unroll
  for (int r = 0; r < 8; ++r) {
    const int    m   = m0 + r + 8 * lhalf;
    const size_t idx = (size_t)m * UNITS + ucol;
    const float iv = sigm(acc[0][r] + bi);
    const float fv = sigm(acc[1][r] + bff);
    const float gv = tanhf(acc[2][r] + bg);
    const float ov = sigm(acc[3][r] + bo);
    const float cn = fv * cst[idx] + iv * gv;
    cst[idx] = cn;
    hbout[idx] = f2bf(ov * tanhf(cn));
  }
}

// ---------------------------------------------------------------------------
// Classifier: out[b] = sigmoid( (h@W1 + b1) @ W2 + b2 )   (33 MFLOP — trivial)
// ---------------------------------------------------------------------------
__global__ void classifier_kernel(const u16* __restrict__ hb,
                                  const float* __restrict__ W1,
                                  const float* __restrict__ b1,
                                  const float* __restrict__ W2,
                                  const float* __restrict__ b2,
                                  float* __restrict__ out) {
  __shared__ float red[64];
  const int b = blockIdx.x;    // BATCH
  const int j = threadIdx.x;   // 64
  const u16* hr = hb + (size_t)b * UNITS;
  float acc = b1[j];
  for (int k = 0; k < UNITS; ++k)
    acc = fmaf(bf2f(hr[k]), W1[k * 64 + j], acc);
  red[j] = acc * W2[j];
  __syncthreads();
  if (j == 0) {
    float s = b2[0];
#pragma unroll
    for (int k = 0; k < 64; ++k) s += red[k];
    out[b] = sigm(s);
  }
}

// ---------------------------------------------------------------------------
extern "C" void kernel_launch(void* const* d_in, const int* in_sizes, int n_in,
                              void* d_out, int out_size, void* d_ws, size_t ws_size,
                              hipStream_t stream) {
  const int*   sents = (const int*)  d_in[0];
  const float* emb   = (const float*)d_in[1];
  const int    vocab_elems = in_sizes[1];        // VOCAB * EMBD
  const float* Wx    = (const float*)d_in[2];    // [512, 4096]
  const float* Wh    = (const float*)d_in[3];    // [1024, 4096]
  const float* bvec  = (const float*)d_in[4];    // [4096]
  const float* W1    = (const float*)d_in[5];    // [1024, 64]
  const float* b1    = (const float*)d_in[6];    // [64]
  const float* W2    = (const float*)d_in[7];    // [64, 1]
  const float* b2    = (const float*)d_in[8];    // [1]
  float* out = (float*)d_out;

  // Workspace layout: small fixed buffers first, bf16 emb table at the tail.
  const size_t WXT_OFF = 0;                                    // 4 MB
  const size_t WHT_OFF = WXT_OFF + (size_t)4 * UNITS * EMBD * 2;   // 8 MB
  const size_t CST_OFF = WHT_OFF + (size_t)4 * UNITS * UNITS * 2;  // 1 MB
  const size_t HB0_OFF = CST_OFF + (size_t)BATCH * UNITS * 4;      // 0.5 MB
  const size_t HB1_OFF = HB0_OFF + (size_t)BATCH * UNITS * 2;      // 0.5 MB
  const size_t EMB_OFF = HB1_OFF + (size_t)BATCH * UNITS * 2;
  const size_t TOTAL   = EMB_OFF + (size_t)vocab_elems * 2;

  char* ws = (char*)d_ws;
  u16*   WxT  = (u16*)(ws + WXT_OFF);
  u16*   WhT  = (u16*)(ws + WHT_OFF);
  float* cst  = (float*)(ws + CST_OFF);
  u16*   hb0  = (u16*)(ws + HB0_OFF);
  u16*   hb1  = (u16*)(ws + HB1_OFF);
  u16*   embb = (u16*)(ws + EMB_OFF);

  const bool preconv = (ws_size >= TOTAL);   // constant per process -> deterministic

  // One-time weight transpose/convert to bf16 (then L2-resident).
  transpose_bf16_kernel<<<(EMBD * 4 * UNITS + 255) / 256, 256, 0, stream>>>(
      Wx, WxT, EMBD, 4 * UNITS);
  transpose_bf16_kernel<<<(UNITS * 4 * UNITS + 255) / 256, 256, 0, stream>>>(
      Wh, WhT, UNITS, 4 * UNITS);
  init_state_kernel<<<(BATCH * UNITS) / 256, 256, 0, stream>>>(cst, hb0);
  if (preconv) {
    convert_bf16_kernel<<<(vocab_elems + 255) / 256, 256, 0, stream>>>(
        emb, embb, vocab_elems);
  }

  // Sequential recurrence: one fused kernel per step; launch boundary = h sync.
  for (int t = 0; t < SEQL; ++t) {
    const u16* hin  = (t & 1) ? hb1 : hb0;
    u16*       hout = (t & 1) ? hb0 : hb1;
    if (preconv) {
      lstm_step_kernel<true><<<128, 256, 0, stream>>>(
          sents, emb, embb, WxT, WhT, bvec, cst, hin, hout, t);
    } else {
      lstm_step_kernel<false><<<128, 256, 0, stream>>>(
          sents, emb, embb, WxT, WhT, bvec, cst, hin, hout, t);
    }
  }
  // T=256 (even) -> final h is in hb0.
  classifier_kernel<<<BATCH, 64, 0, stream>>>(hb0, W1, b1, W2, b2, out);
}
